// HierarchicalGraphNeuralNetwork_56032143344105
// MI455X (gfx1250) — compile-verified, matrix operand order checked by reading, not run on hardware
//
#include <hip/hip_runtime.h>
#include <math.h>

// ---------------- CDNA5 WMMA types ----------------
typedef __attribute__((ext_vector_type(16))) __bf16 v16bf;              // 8 VGPRs
typedef __attribute__((ext_vector_type(8)))  float  v8f;                // 8 VGPRs
typedef __attribute__((ext_vector_type(8), aligned(16))) __bf16 v8bf16; // 16B chunk
typedef __attribute__((ext_vector_type(4)))  float  f32x4;              // 16B aligned

// Problem constants (from reference)
#define BATCH      64
#define F_PER      256
#define NPF        32
#define E_EXT      64
#define D_HID      64
#define D_IN       11
#define D_IN_PAD   16
#define N_CFG      (BATCH * F_PER * NPF)      // 524288
#define E_CFG_N    (4LL * N_CFG)              // 2097152
#define N_FCG_PER  (F_PER + E_EXT)            // 320
#define N_FCG      (BATCH * N_FCG_PER)        // 20480
#define E_FCG_N    (8LL * N_FCG)              // 163840

// ---------------- utility kernels ----------------
__global__ void zero_kernel(float* __restrict__ p, long long n) {
  long long stride = (long long)gridDim.x * blockDim.x;
  for (long long i = (long long)blockIdx.x * blockDim.x + threadIdx.x; i < n; i += stride)
    p[i] = 0.0f;
}

// rows x 11 -> rows x 16, zero padded
__global__ void pad_x_kernel(const float* __restrict__ in, float* __restrict__ outp,
                             long long nRows) {
  long long total = nRows * D_IN_PAD;
  long long stride = (long long)gridDim.x * blockDim.x;
  for (long long i = (long long)blockIdx.x * blockDim.x + threadIdx.x; i < total; i += stride) {
    long long row = i >> 4;
    int c = (int)(i & 15);
    outp[i] = (c < D_IN) ? in[row * D_IN + c] : 0.0f;
  }
}

__global__ void edge_count_kernel(const int* __restrict__ ei, long long E,
                                  float* __restrict__ cnt) {
  long long stride = (long long)gridDim.x * blockDim.x;
  for (long long e = (long long)blockIdx.x * blockDim.x + threadIdx.x; e < E; e += stride) {
    atomicAdd(&cnt[ei[E + e]], 1.0f);
  }
}

// msg[dst, c] += x[src, c]  for c in [0, D); x stride ld, msg stride msgLd
__global__ void edge_scatter_kernel(const int* __restrict__ ei, long long E, int D,
                                    const float* __restrict__ x, int ld,
                                    float* __restrict__ msg, int msgLd) {
  long long total = E * (long long)D;
  long long stride = (long long)gridDim.x * blockDim.x;
  for (long long i = (long long)blockIdx.x * blockDim.x + threadIdx.x; i < total; i += stride) {
    long long e = i / D;
    int c = (int)(i - e * D);
    int s = ei[e];
    int d = ei[E + e];
    atomicAdd(&msg[(long long)d * msgLd + c], x[(long long)s * ld + c]);
  }
}

__global__ void mean_div_kernel(float* __restrict__ msg, const float* __restrict__ cnt,
                                long long N, int D) {
  long long total = N * (long long)D;
  long long stride = (long long)gridDim.x * blockDim.x;
  for (long long i = (long long)blockIdx.x * blockDim.x + threadIdx.x; i < total; i += stride) {
    msg[i] = msg[i] / fmaxf(cnt[i / D], 1.0f);
  }
}

// ---------------- fused SAGE layer: Out = relu(X@Wl + M@Wr + b) via WMMA ----------------
// A-fragment element map (16x32 bf16, ISA layout):
//   element e -> k = kk + (e>>3)*16 + (lane>>4)*8 + (e&7)
// KIN is a multiple of 16, so each octet is fully in-range or fully zero (g-uniform).
template <int KIN>
__device__ inline v16bf load_a_frag(const float* __restrict__ S, int rowBase, int kk,
                                    int lane) {
  int half = (lane >> 4) & 1;
  int m    = lane & 15;
  const float* rowp = S + (long long)(rowBase + m) * KIN;
  v16bf a;
#pragma unroll
  for (int g = 0; g < 2; ++g) {
    int kfirst = kk + g * 16;  // compile-time after unroll
    if (kfirst < KIN) {
      int kbase = kfirst + half * 8;
      f32x4 lo = *(const f32x4*)(rowp + kbase);      // global_load_b128
      f32x4 hi = *(const f32x4*)(rowp + kbase + 4);  // global_load_b128
#pragma unroll
      for (int j = 0; j < 4; ++j) {
        a[g * 8 + j]     = (__bf16)lo[j];
        a[g * 8 + 4 + j] = (__bf16)hi[j];
      }
    } else {
#pragma unroll
      for (int j = 0; j < 8; ++j) a[g * 8 + j] = (__bf16)0.0f;
    }
  }
  return a;
}

// B fragments pre-swizzled in LDS: chunk ((ks*4 + nT)*32 + lane) holds that lane's
// 16 contiguous bf16 elements (32B, aligned) -> one v16bf load (2x ds_load_b128).
__device__ inline v16bf load_b_frag(const __bf16* __restrict__ Wswz, int ks, int nT,
                                    int lane) {
  return *(const v16bf*)(Wswz + ((((ks << 2) + nT) << 5) + lane) * 16);
}

// block = 128 threads = 4 waves; each wave -> 2 row-tiles (32 rows); block -> 128 rows.
// Safe to alias Out with M: each wave reads all of its rows before storing them,
// and rows are partitioned exclusively across waves.
// KIN: padded feature stride (16 or 64); KTRUE: true weight rows (11 or 64).
template <int KIN, int KTRUE>
__global__ __launch_bounds__(128) void sage_wmma_kernel(
    const float* __restrict__ X, const float* __restrict__ M,
    const float* __restrict__ Wl, const float* __restrict__ Wr,
    const float* __restrict__ bias, float* __restrict__ Out, int nRows) {
  constexpr int KPAD = (KIN + 31) & ~31;  // 32 or 64
  __shared__ __align__(32) __bf16 wl_s[KPAD * D_HID];  // swizzled fragment order
  __shared__ __align__(32) __bf16 wr_s[KPAD * D_HID];

  // Stage weights pre-swizzled; one 16B octet per iteration (ds_store_b128).
  // Flat octet index i: oct=i&1, lane=(i>>1)&31, nT=(i>>6)&3, ks=i>>8.
  constexpr int totalOct = KPAD * 8;
#pragma unroll 2
  for (int i = threadIdx.x; i < totalOct; i += 128) {
    int oct  = i & 1;
    int lane = (i >> 1) & 31;
    int nT   = (i >> 6) & 3;
    int ks   = i >> 8;
    int half = lane >> 4;
    int n    = nT * 16 + (lane & 15);
    int kbase = ks * 32 + oct * 16 + half * 8;
    v8bf16 vl, vr;
#pragma unroll
    for (int j = 0; j < 8; ++j) {
      int k = kbase + j;
      vl[j] = (__bf16)((k < KTRUE) ? Wl[k * D_HID + n] : 0.0f);
      vr[j] = (__bf16)((k < KTRUE) ? Wr[k * D_HID + n] : 0.0f);
    }
    *(v8bf16*)(wl_s + (long long)i * 8) = vl;
    *(v8bf16*)(wr_s + (long long)i * 8) = vr;
  }
  __syncthreads();

  int wave = threadIdx.x >> 5;
  int lane = threadIdx.x & 31;
  int rowBase = (blockIdx.x * 4 + wave) * 32;
  if (rowBase >= nRows) return;  // wave-uniform: EXEC stays all-1s for WMMA

  v8f zero = {0.f, 0.f, 0.f, 0.f, 0.f, 0.f, 0.f, 0.f};
  v8f acc[2][4];
#pragma unroll
  for (int rt = 0; rt < 2; ++rt)
#pragma unroll
    for (int t = 0; t < 4; ++t) acc[rt][t] = zero;

#pragma unroll
  for (int kk = 0; kk < KPAD; kk += 32) {
    int ks = kk >> 5;
    v16bf aX0 = load_a_frag<KIN>(X, rowBase,      kk, lane);
    v16bf aM0 = load_a_frag<KIN>(M, rowBase,      kk, lane);
    v16bf aX1 = load_a_frag<KIN>(X, rowBase + 16, kk, lane);
    v16bf aM1 = load_a_frag<KIN>(M, rowBase + 16, kk, lane);
#pragma unroll
    for (int nT = 0; nT < 4; ++nT) {
      v16bf bL = load_b_frag(wl_s, ks, nT, lane);
      v16bf bR = load_b_frag(wr_s, ks, nT, lane);
      acc[0][nT] = __builtin_amdgcn_wmma_f32_16x16x32_bf16(
          false, aX0, false, bL, (short)0, acc[0][nT], false, false);
      acc[0][nT] = __builtin_amdgcn_wmma_f32_16x16x32_bf16(
          false, aM0, false, bR, (short)0, acc[0][nT], false, false);
      acc[1][nT] = __builtin_amdgcn_wmma_f32_16x16x32_bf16(
          false, aX1, false, bL, (short)0, acc[1][nT], false, false);
      acc[1][nT] = __builtin_amdgcn_wmma_f32_16x16x32_bf16(
          false, aM1, false, bR, (short)0, acc[1][nT], false, false);
    }
  }

  // C/D layout: VGPR r -> M = r (+8 for lanes 16-31); lane&15 -> N
  int half = lane >> 4;
  int nlo  = lane & 15;
#pragma unroll
  for (int rt = 0; rt < 2; ++rt) {
#pragma unroll
    for (int nT = 0; nT < 4; ++nT) {
#pragma unroll
      for (int r = 0; r < 8; ++r) {
        int row = rowBase + rt * 16 + half * 8 + r;
        int col = nT * 16 + nlo;
        float v = acc[rt][nT][r] + bias[col];
        Out[(long long)row * D_HID + col] = v > 0.0f ? v : 0.0f;
      }
    }
  }
}

// ---------------- fixed-size segment max pool ----------------
__global__ void segmax_kernel(const float* __restrict__ h, int segSize, int numSeg,
                              float* __restrict__ out) {
  int idx = blockIdx.x * blockDim.x + threadIdx.x;  // seg*64 + col
  if (idx >= numSeg * D_HID) return;
  int seg = idx >> 6;
  int col = idx & 63;
  const float* base = h + (long long)seg * segSize * D_HID + col;
  float m = -INFINITY;
  for (int r = 0; r < segSize; ++r) m = fmaxf(m, base[(long long)r * D_HID]);
  out[idx] = m;
}

// ---------------- FCG feature assembly: [internal | external emb] ----------------
__global__ void assemble_fcg_kernel(const float* __restrict__ pooled,
                                    const float* __restrict__ emb,
                                    const int* __restrict__ ids,
                                    float* __restrict__ fx) {
  int idx = blockIdx.x * blockDim.x + threadIdx.x;  // node*64 + col
  if (idx >= N_FCG * D_HID) return;
  int node = idx >> 6;
  int col  = idx & 63;
  int sample = node / N_FCG_PER;
  int loc    = node - sample * N_FCG_PER;
  float v;
  if (loc < F_PER) {
    v = pooled[((long long)sample * F_PER + loc) * D_HID + col];
  } else {
    int id = ids[sample * E_EXT + (loc - F_PER)];
    v = emb[(long long)id * D_HID + col];
  }
  fx[idx] = v;
}

// ---------------- projection head: sigmoid(pj3(pj2(pj1(x)))) ----------------
__global__ void head_kernel(const float* __restrict__ pooled,
                            const float* __restrict__ w1, const float* __restrict__ b1,
                            const float* __restrict__ w2, const float* __restrict__ b2,
                            const float* __restrict__ w3, const float* __restrict__ b3,
                            float* __restrict__ out) {
  int s = threadIdx.x;
  if (s >= BATCH) return;
  const float* p = pooled + s * D_HID;
  float h1[32];
  for (int j = 0; j < 32; ++j) {
    float sum = b1[j];
    for (int i = 0; i < 64; ++i) sum += p[i] * w1[i * 32 + j];
    h1[j] = sum;
  }
  float h2[16];
  for (int j = 0; j < 16; ++j) {
    float sum = b2[j];
    for (int i = 0; i < 32; ++i) sum += h1[i] * w2[i * 16 + j];
    h2[j] = sum;
  }
  float o = b3[0];
  for (int i = 0; i < 16; ++i) o += h2[i] * w3[i];
  out[s] = 1.0f / (1.0f + expf(-o));
}

// ---------------- host orchestration ----------------
extern "C" void kernel_launch(void* const* d_in, const int* in_sizes, int n_in,
                              void* d_out, int out_size, void* d_ws, size_t ws_size,
                              hipStream_t stream) {
  (void)in_sizes; (void)n_in; (void)out_size; (void)ws_size;

  const float* cfg_x    = (const float*)d_in[0];
  const float* cfg_w1l  = (const float*)d_in[1];
  const float* cfg_w1r  = (const float*)d_in[2];
  const float* cfg_b1   = (const float*)d_in[3];
  const float* cfg_w2l  = (const float*)d_in[4];
  const float* cfg_w2r  = (const float*)d_in[5];
  const float* cfg_b2   = (const float*)d_in[6];
  const float* emb      = (const float*)d_in[7];
  const float* fcg_w1l  = (const float*)d_in[8];
  const float* fcg_w1r  = (const float*)d_in[9];
  const float* fcg_b1   = (const float*)d_in[10];
  const float* fcg_w2l  = (const float*)d_in[11];
  const float* fcg_w2r  = (const float*)d_in[12];
  const float* fcg_b2   = (const float*)d_in[13];
  const float* pj1_w    = (const float*)d_in[14];
  const float* pj1_b    = (const float*)d_in[15];
  const float* pj2_w    = (const float*)d_in[16];
  const float* pj2_b    = (const float*)d_in[17];
  const float* pj3_w    = (const float*)d_in[18];
  const float* pj3_b    = (const float*)d_in[19];
  const int*   cfg_ei   = (const int*)d_in[20];
  const int*   ext_ids  = (const int*)d_in[22];
  const int*   fcg_ei   = (const int*)d_in[23];
  float* out = (float*)d_out;

  // workspace carve-out (floats, 256B aligned)
  uint8_t* base = (uint8_t*)d_ws;
  size_t cur = 0;
  auto carve = [&](long long nFloats) -> float* {
    float* p = (float*)(base + cur);
    cur += ((size_t)nFloats * sizeof(float) + 255) & ~(size_t)255;
    return p;
  };
  float* cnt_cfg    = carve(N_CFG);                          //   2 MB
  float* x16        = carve((long long)N_CFG * D_IN_PAD);    //  32 MB (padded cfg_x)
  float* msg1       = carve((long long)N_CFG * D_IN_PAD);    //  32 MB (padded mean)
  float* h1         = carve((long long)N_CFG * D_HID);       // 128 MB
  float* msg2       = carve((long long)N_CFG * D_HID);       // 128 MB (h2 in-place)
  float* pooled_cfg = carve((long long)BATCH * F_PER * D_HID);  // 4 MB
  float* fcg_x      = carve((long long)N_FCG * D_HID);
  float* cnt_fcg    = carve(N_FCG);
  float* fmsg1      = carve((long long)N_FCG * D_HID);
  float* g1         = carve((long long)N_FCG * D_HID);
  float* fmsg2      = carve((long long)N_FCG * D_HID);       // g2 in-place
  float* pooled_fcg = carve((long long)BATCH * D_HID);

  const int ZB = 2048, SB = 8192, T = 256;

  // ---- CFG layer 1 ----
  zero_kernel<<<ZB, T, 0, stream>>>(cnt_cfg, (long long)N_CFG);
  zero_kernel<<<ZB, T, 0, stream>>>(msg1, (long long)N_CFG * D_IN_PAD);
  pad_x_kernel<<<ZB, T, 0, stream>>>(cfg_x, x16, (long long)N_CFG);
  edge_count_kernel<<<SB, T, 0, stream>>>(cfg_ei, E_CFG_N, cnt_cfg);
  edge_scatter_kernel<<<SB, T, 0, stream>>>(cfg_ei, E_CFG_N, D_IN, cfg_x, D_IN,
                                            msg1, D_IN_PAD);
  mean_div_kernel<<<ZB, T, 0, stream>>>(msg1, cnt_cfg, (long long)N_CFG, D_IN_PAD);
  sage_wmma_kernel<D_IN_PAD, D_IN><<<N_CFG / 128, 128, 0, stream>>>(
      x16, msg1, cfg_w1l, cfg_w1r, cfg_b1, h1, N_CFG);
  // ---- CFG layer 2 ----
  zero_kernel<<<ZB, T, 0, stream>>>(msg2, (long long)N_CFG * D_HID);
  edge_scatter_kernel<<<SB, T, 0, stream>>>(cfg_ei, E_CFG_N, D_HID, h1, D_HID,
                                            msg2, D_HID);
  mean_div_kernel<<<ZB, T, 0, stream>>>(msg2, cnt_cfg, (long long)N_CFG, D_HID);
  sage_wmma_kernel<D_HID, D_HID><<<N_CFG / 128, 128, 0, stream>>>(
      h1, msg2, cfg_w2l, cfg_w2r, cfg_b2, msg2, N_CFG);
  // ---- per-function max pool + FCG assembly ----
  segmax_kernel<<<(BATCH * F_PER * D_HID + T - 1) / T, T, 0, stream>>>(
      msg2, NPF, BATCH * F_PER, pooled_cfg);
  assemble_fcg_kernel<<<(N_FCG * D_HID + T - 1) / T, T, 0, stream>>>(
      pooled_cfg, emb, ext_ids, fcg_x);

  // ---- FCG layer 1 ----
  zero_kernel<<<64, T, 0, stream>>>(cnt_fcg, (long long)N_FCG);
  zero_kernel<<<512, T, 0, stream>>>(fmsg1, (long long)N_FCG * D_HID);
  edge_count_kernel<<<512, T, 0, stream>>>(fcg_ei, E_FCG_N, cnt_fcg);
  edge_scatter_kernel<<<2048, T, 0, stream>>>(fcg_ei, E_FCG_N, D_HID, fcg_x, D_HID,
                                              fmsg1, D_HID);
  mean_div_kernel<<<512, T, 0, stream>>>(fmsg1, cnt_fcg, (long long)N_FCG, D_HID);
  sage_wmma_kernel<D_HID, D_HID><<<N_FCG / 128, 128, 0, stream>>>(
      fcg_x, fmsg1, fcg_w1l, fcg_w1r, fcg_b1, g1, N_FCG);
  // ---- FCG layer 2 ----
  zero_kernel<<<512, T, 0, stream>>>(fmsg2, (long long)N_FCG * D_HID);
  edge_scatter_kernel<<<2048, T, 0, stream>>>(fcg_ei, E_FCG_N, D_HID, g1, D_HID,
                                              fmsg2, D_HID);
  mean_div_kernel<<<512, T, 0, stream>>>(fmsg2, cnt_fcg, (long long)N_FCG, D_HID);
  sage_wmma_kernel<D_HID, D_HID><<<N_FCG / 128, 128, 0, stream>>>(
      g1, fmsg2, fcg_w2l, fcg_w2r, fcg_b2, fmsg2, N_FCG);
  // ---- per-sample max pool + head ----
  segmax_kernel<<<(BATCH * D_HID + T - 1) / T, T, 0, stream>>>(
      fmsg2, N_FCG_PER, BATCH, pooled_fcg);
  head_kernel<<<1, 64, 0, stream>>>(pooled_fcg, pj1_w, pj1_b, pj2_w, pj2_b,
                                    pj3_w, pj3_b, out);
}
